// SegGRU_72284299592179
// MI455X (gfx1250) — matrix operand order repaired and checked by reading, not compiled, not run on hardware
//
#include <hip/hip_runtime.h>
#include <hip/hip_bf16.h>

#define T_SEQ 40
#define H_DIM 128
#define C_CLS 102

typedef __attribute__((ext_vector_type(16))) __bf16 v16bf;
typedef __attribute__((ext_vector_type(8)))  float  v8f;
typedef unsigned int u32x4 __attribute__((ext_vector_type(4)));
typedef int          i32x8 __attribute__((ext_vector_type(8)));
typedef int          i32x4 __attribute__((ext_vector_type(4)));

#if __has_builtin(__builtin_amdgcn_tensor_load_to_lds) && __has_builtin(__builtin_amdgcn_s_wait_tensorcnt)
#define USE_TDM 1
#endif

__device__ __forceinline__ v8f wmma_bf16(v16bf a, v16bf b, v8f c){
  return __builtin_amdgcn_wmma_f32_16x16x32_bf16(false, a, false, b, (short)0, c, false, false);
}

__device__ __forceinline__ v8f splat8(float x){
  v8f r;
#pragma unroll
  for (int i = 0; i < 8; ++i) r[i] = x;
  return r;
}

__device__ __forceinline__ float fast_sigmoid(float x){
  return __builtin_amdgcn_rcpf(1.f + __expf(-x));
}
__device__ __forceinline__ float fast_tanh(float x){
  float e = __expf(2.f * x);                       // inf-safe: e=inf -> 1, e=0 -> -1
  return 1.f - 2.f * __builtin_amdgcn_rcpf(e + 1.f);
}

// A fragment (16x32 bf16, M x K). base is row-major [rows][ld], k contiguous.
// lane 0-15: M=lane, K 0..15; lane 16-31: M=lane-16, K 16..31 (contiguous).
__device__ __forceinline__ v16bf load_A_rm(const __bf16* __restrict__ base, int row0, int ld,
                                           int kbase, int lane){
  union { v16bf v; unsigned u[8]; } f;
  int m  = lane & 15;
  int hh = lane >> 4;
  const __bf16* rp = base + (row0 + m) * ld + kbase + hh * 16;
#pragma unroll
  for (int j = 0; j < 8; ++j){
    int k = j << 1;
    f.u[j] = *(const unsigned*)(rp + k);
  }
  return f.v;
}

// B fragment (32x16, K x N). Element (k,n) at base[n*ld + k] (k contiguous per n).
__device__ __forceinline__ v16bf load_B_nk(const __bf16* __restrict__ base, int n0, int ld,
                                           int kbase, int lane){
  union { v16bf v; unsigned u[8]; } f;
  int n  = n0 + (lane & 15);
  int kb = kbase + ((lane >> 4) << 4);
  const __bf16* rp = base + n * ld + kb;
#pragma unroll
  for (int j = 0; j < 8; ++j) f.u[j] = *(const unsigned*)(rp + (j << 1));
  return f.v;
}

// B fragment (32x16, K x N). Element (k,n) at base[k*ld + n] (gathered 16-bit).
__device__ __forceinline__ v16bf load_B_kn(const __bf16* __restrict__ base, int n0, int ld,
                                           int kbase, int lane){
  union { v16bf v; __bf16 h[16]; } f;
  int n  = n0 + (lane & 15);
  int kb = kbase + ((lane >> 4) << 4);
#pragma unroll
  for (int j = 0; j < 16; ++j) f.h[j] = base[(kb + j) * ld + n];
  return f.v;
}

#ifdef USE_TDM
// TDM: DMA a contiguous run of `nelem` bf16 elements (1-D tile) from global to LDS.
// D# built per CDNA5 ISA ch.8: group0 = {count|flags, lds_addr, gaddr_lo, gaddr_hi|type=2},
// group1 = {data_size=2B, tensor_dim0/1, tile_dim0/1, dim0_stride}; groups 2/3 zero (<=2D).
__device__ __forceinline__ void tdm_load_bf16_1d(unsigned lds_addr, const void* gptr, unsigned nelem){
  unsigned long long ga = (unsigned long long)(size_t)gptr;
  u32x4 g0;
  g0[0] = 1u;                                               // count=1, user descriptor
  g0[1] = lds_addr;                                         // LDS byte address
  g0[2] = (unsigned)(ga & 0xFFFFFFFFu);                     // global_addr[31:0]
  g0[3] = (unsigned)((ga >> 32) & 0x01FFFFFFu) | 0x80000000u; // global_addr[56:32] | type=2
  i32x8 g1;
  g1[0] = (1 << 16);                                        // workgroup_mask=0, data_size=2B
  g1[1] = (int)((nelem & 0xFFFFu) << 16);                   // tensor_dim0[15:0]
  g1[2] = (int)(((nelem >> 16) & 0xFFFFu) | (1u << 16));    // tensor_dim0[31:16], tensor_dim1=1
  g1[3] = (int)((nelem & 0xFFFFu) << 16);                   // tile_dim0 = nelem
  g1[4] = 1;                                                // tile_dim1 = 1
  g1[5] = (int)nelem;                                       // tensor_dim0_stride
  g1[6] = 0;
  g1[7] = 0;
  i32x4 z4 = {0, 0, 0, 0};
#if __clang_major__ >= 23
  i32x8 z8 = {0, 0, 0, 0, 0, 0, 0, 0};
  __builtin_amdgcn_tensor_load_to_lds(g0, g1, z4, z4, z8, 0);
#else
  __builtin_amdgcn_tensor_load_to_lds(g0, g1, z4, z4, 0);
#endif
}
#endif

// ---------------- prep: fp32 -> bf16 weight conversion ----------------
__global__ __launch_bounds__(256) void prep_kernel(
    const float* __restrict__ W_ih, const float* __restrict__ W_hh,
    const float* __restrict__ W_cls,
    __bf16* __restrict__ Wih_bf, __bf16* __restrict__ Whh_bf,
    __bf16* __restrict__ Wcls_bf){
  int i = blockIdx.x * 256 + threadIdx.x;
  if (i < 3 * H_DIM * H_DIM){
    Wih_bf[i] = (__bf16)W_ih[i];
    Whh_bf[i] = (__bf16)W_hh[i];
  }
  if (i < 112 * H_DIM){
    int r = i >> 7;
    Wcls_bf[i] = (r < C_CLS) ? (__bf16)W_cls[i] : (__bf16)0.f;
  }
}

// ---------------- fused embed-gather + GRU recurrence ----------------
// grid.x = B/16 batch tiles, grid.y = {0: emb_a->hs_a, 1: emb_c->hs_c}
// 256 threads = 8 waves; wave w owns hidden columns [16w, 16w+16).
__global__ __launch_bounds__(256) void gru_kernel(
    const int* __restrict__ ids,
    const float* __restrict__ emb_a, const float* __restrict__ emb_c,
    const float* __restrict__ b_ih,  const float* __restrict__ b_hh,
    const __bf16* __restrict__ Wih_bf, const __bf16* __restrict__ Whh_bf,
    __bf16* __restrict__ hs_a, __bf16* __restrict__ hs_c){
  const int tid   = threadIdx.x;
  const int lane  = tid & 31;
  const int wave  = tid >> 5;
  const int bbase = blockIdx.x * 16;
  const float* __restrict__ emb = (blockIdx.y == 0) ? emb_a : emb_c;
  __bf16* __restrict__ hs       = (blockIdx.y == 0) ? hs_a  : hs_c;

  __shared__ __align__(16) __bf16 x_lds[16 * H_DIM];
  __shared__ __align__(16) __bf16 h_lds[16 * H_DIM];

  for (int i = tid; i < 16 * H_DIM; i += 256) h_lds[i] = (__bf16)0.f;   // h0 = 0

  const int colg = wave * 16 + (lane & 15);
  const float vr  = b_ih[colg]             + b_hh[colg];
  const float vz  = b_ih[H_DIM + colg]     + b_hh[H_DIM + colg];
  const float vin = b_ih[2 * H_DIM + colg];
  const float vhn = b_hh[2 * H_DIM + colg];

  const int gm = tid >> 4;          // 0..15 : row for gather / writeback
  const int gk = (tid & 15) * 8;    // 0..120: 8 hidden elems per thread

  for (int t = 0; t < T_SEQ; ++t){
    __syncthreads();                               // prior-step LDS reads done
    {
      int row = ids[(bbase + gm) * T_SEQ + t];     // embedding gather (L2-hot)
      const float* src = emb + (long)row * H_DIM + gk;
#pragma unroll
      for (int j = 0; j < 8; ++j) x_lds[gm * H_DIM + gk + j] = (__bf16)src[j];
      if (t + 1 < T_SEQ){                          // hide next-step gather latency
        int rown = ids[(bbase + gm) * T_SEQ + t + 1];
        __builtin_prefetch(emb + (long)rown * H_DIM + gk, 0, 3);  // global_prefetch_b8
      }
    }
    __syncthreads();                               // x_lds + h_lds ready

    v8f ar = splat8(vr), az = splat8(vz), ain = splat8(vin), ahn = splat8(vhn);
#pragma unroll
    for (int c = 0; c < 4; ++c){
      const int kb = c * 32;
      v16bf axf = load_A_rm(x_lds, 0, H_DIM, kb, lane);
      v16bf ahf = load_A_rm(h_lds, 0, H_DIM, kb, lane);
      v16bf w;
      w = load_B_nk(Wih_bf, wave * 16,             H_DIM, kb, lane); ar  = wmma_bf16(axf, w, ar);
      w = load_B_nk(Whh_bf, wave * 16,             H_DIM, kb, lane); ar  = wmma_bf16(ahf, w, ar);
      w = load_B_nk(Wih_bf, H_DIM + wave * 16,     H_DIM, kb, lane); az  = wmma_bf16(axf, w, az);
      w = load_B_nk(Whh_bf, H_DIM + wave * 16,     H_DIM, kb, lane); az  = wmma_bf16(ahf, w, az);
      w = load_B_nk(Wih_bf, 2 * H_DIM + wave * 16, H_DIM, kb, lane); ain = wmma_bf16(axf, w, ain);
      w = load_B_nk(Whh_bf, 2 * H_DIM + wave * 16, H_DIM, kb, lane); ahn = wmma_bf16(ahf, w, ahn);
    }

    // gate math in the C-layout: lane n = colg, VGPR i -> M = i + 8*(lane>>4)
    float hnew[8];
#pragma unroll
    for (int i = 0; i < 8; ++i){
      int m   = i + ((lane >> 4) << 3);
      float r = fast_sigmoid(ar[i]);
      float z = fast_sigmoid(az[i]);
      float n = fast_tanh(ain[i] + r * ahn[i]);
      float ho = (float)h_lds[m * H_DIM + colg];
      hnew[i] = (1.f - z) * n + z * ho;
    }
    __syncthreads();                               // all h_lds reads done
#pragma unroll
    for (int i = 0; i < 8; ++i){
      int m = i + ((lane >> 4) << 3);
      h_lds[m * H_DIM + colg] = (__bf16)hnew[i];
    }
    __syncthreads();                               // h_t complete
    {
      __bf16* dst = hs + ((long)(bbase + gm) * T_SEQ + t) * H_DIM + gk;
      *(uint4*)dst = *(const uint4*)(h_lds + gm * H_DIM + gk);   // coalesced b128
    }
  }
}

// ---------------- attention (symmetric Gram) + classifier ----------------
// one block per batch element; 256 threads = 8 waves
__global__ __launch_bounds__(256) void attn_cls_kernel(
    const __bf16* __restrict__ hs_a, const __bf16* __restrict__ hs_c,
    const __bf16* __restrict__ Wcls_bf, const float* __restrict__ b_cls,
    float* __restrict__ out){
  const int b    = blockIdx.x;
  const int tid  = threadIdx.x;
  const int lane = tid & 31;
  const int wave = tid >> 5;

  // aliased LDS regions (56320 B total):
  //   [0,16384)       hc_s  : bf16 [64][128]   (K-padded rows zeroed)
  //   [16384,22528)   P_s   : bf16 [48][64]
  //   [22528,34816)   ha_s  : bf16 [48][128] -> reused as o_s after GEMM1 barrier
  //   [34816,56320)   m_s   : f32  [48][48]  -> reused as l_s f32 [48][112]
  __shared__ __align__(16) unsigned char smem[56320];
  __bf16* hc_s = (__bf16*)smem;
  __bf16* P_s  = (__bf16*)(smem + 16384);
  __bf16* ha_s = (__bf16*)(smem + 22528);
  __bf16* o_s  = ha_s;
  float*  m_s  = (float*)(smem + 34816);
  float*  l_s  = m_s;

  // zero only the M/K pad rows (rows >= 40); valid rows come via TDM / copy below
  for (int i = tid + T_SEQ * H_DIM; i < 48 * H_DIM; i += 256) ha_s[i] = (__bf16)0.f;
  for (int i = tid + T_SEQ * H_DIM; i < 64 * H_DIM; i += 256) hc_s[i] = (__bf16)0.f;

#ifdef USE_TDM
  if (tid == 0){
    // Tensor Data Mover: async DMA of the two 40x128 bf16 tiles straight into LDS
    tdm_load_bf16_1d((unsigned)(size_t)(void*)ha_s,
                     hs_a + (long)b * T_SEQ * H_DIM, T_SEQ * H_DIM);
    tdm_load_bf16_1d((unsigned)(size_t)(void*)hc_s,
                     hs_c + (long)b * T_SEQ * H_DIM, T_SEQ * H_DIM);
    __builtin_amdgcn_s_wait_tensorcnt(0);
  }
#else
  for (int i = tid; i < T_SEQ * H_DIM; i += 256) ha_s[i] = hs_a[(long)b * T_SEQ * H_DIM + i];
  for (int i = tid; i < T_SEQ * H_DIM; i += 256) hc_s[i] = hs_c[(long)b * T_SEQ * H_DIM + i];
#endif
  __syncthreads();

  // GEMM1: m[i][j] = ha_i . ha_j   (48x48, K=128)
  for (int tile = wave; tile < 9; tile += 8){
    int ti = tile / 3, tj = tile % 3;
    v8f acc = splat8(0.f);
#pragma unroll
    for (int c = 0; c < 4; ++c){
      v16bf a  = load_A_rm(ha_s, ti * 16, H_DIM, c * 32, lane);
      v16bf bb = load_B_nk(ha_s, tj * 16, H_DIM, c * 32, lane);
      acc = wmma_bf16(a, bb, acc);
    }
    int n = tj * 16 + (lane & 15);
#pragma unroll
    for (int i = 0; i < 8; ++i){
      int mr = ti * 16 + i + ((lane >> 4) << 3);
      m_s[mr * 48 + n] = acc[i];
    }
  }
  __syncthreads();

  // softmax over dim-1 (first index): P[s][t] = exp(m[s][t]-M(t))/Z(t)
  if (tid < 64){
    int tt = tid;
    if (tt < T_SEQ){
      float mx = -1e30f;
      for (int s = 0; s < T_SEQ; ++s) mx = fmaxf(mx, m_s[s * 48 + tt]);
      float Z = 0.f;
      for (int s = 0; s < T_SEQ; ++s) Z += __expf(m_s[s * 48 + tt] - mx);
      float inv = __builtin_amdgcn_rcpf(Z);
      for (int s = 0; s < 48; ++s)
        P_s[s * 64 + tt] = (s < T_SEQ) ? (__bf16)(__expf(m_s[s * 48 + tt] - mx) * inv)
                                       : (__bf16)0.f;
    } else {
      for (int s = 0; s < 48; ++s) P_s[s * 64 + tt] = (__bf16)0.f;
    }
  }
  __syncthreads();

  // GEMM2: out[48,128] = P[48,64] @ hc[64,128]
  for (int tile = wave; tile < 24; tile += 8){
    int ti = tile / 8, tj = tile % 8;
    v8f acc = splat8(0.f);
#pragma unroll
    for (int c = 0; c < 2; ++c){
      v16bf a  = load_A_rm(P_s, ti * 16, 64, c * 32, lane);
      v16bf bb = load_B_kn(hc_s, tj * 16, H_DIM, c * 32, lane);
      acc = wmma_bf16(a, bb, acc);
    }
    int n = tj * 16 + (lane & 15);
#pragma unroll
    for (int i = 0; i < 8; ++i){
      int mr = ti * 16 + i + ((lane >> 4) << 3);
      o_s[mr * H_DIM + n] = (__bf16)acc[i];
    }
  }
  __syncthreads();

  // GEMM3: logits[48,112] = out[48,128] @ Wcls^T[128,112] + b_cls
  for (int tile = wave; tile < 21; tile += 8){
    int ti = tile / 7, tj = tile % 7;
    int n  = tj * 16 + (lane & 15);
    v8f acc = splat8((n < C_CLS) ? b_cls[n] : 0.f);
#pragma unroll
    for (int c = 0; c < 4; ++c){
      v16bf a  = load_A_rm(o_s, ti * 16, H_DIM, c * 32, lane);
      v16bf bb = load_B_nk(Wcls_bf, tj * 16, H_DIM, c * 32, lane);
      acc = wmma_bf16(a, bb, acc);
    }
#pragma unroll
    for (int i = 0; i < 8; ++i){
      int mr = ti * 16 + i + ((lane >> 4) << 3);
      l_s[mr * 112 + n] = acc[i];
    }
  }
  __syncthreads();

  // row softmax over C=102 (wave32 shuffle reduction), write final output
  for (int s = wave; s < T_SEQ; s += 8){
    float v0 = (lane      < C_CLS) ? l_s[s * 112 + lane]      : -1e30f;
    float v1 = (lane + 32 < C_CLS) ? l_s[s * 112 + lane + 32] : -1e30f;
    float v2 = (lane + 64 < C_CLS) ? l_s[s * 112 + lane + 64] : -1e30f;
    float v3 = (lane + 96 < C_CLS) ? l_s[s * 112 + lane + 96] : -1e30f;
    float mx = fmaxf(fmaxf(v0, v1), fmaxf(v2, v3));
#pragma unroll
    for (int o = 16; o > 0; o >>= 1) mx = fmaxf(mx, __shfl_xor(mx, o, 32));
    float e0 = (lane      < C_CLS) ? __expf(v0 - mx) : 0.f;
    float e1 = (lane + 32 < C_CLS) ? __expf(v1 - mx) : 0.f;
    float e2 = (lane + 64 < C_CLS) ? __expf(v2 - mx) : 0.f;
    float e3 = (lane + 96 < C_CLS) ? __expf(v3 - mx) : 0.f;
    float sum = e0 + e1 + e2 + e3;
#pragma unroll
    for (int o = 16; o > 0; o >>= 1) sum += __shfl_xor(sum, o, 32);
    float inv = __builtin_amdgcn_rcpf(sum);
    float* dst = out + ((long)b * T_SEQ + s) * C_CLS;
    if (lane      < C_CLS) dst[lane]      = e0 * inv;
    if (lane + 32 < C_CLS) dst[lane + 32] = e1 * inv;
    if (lane + 64 < C_CLS) dst[lane + 64] = e2 * inv;
    if (lane + 96 < C_CLS) dst[lane + 96] = e3 * inv;
  }
}

extern "C" void kernel_launch(void* const* d_in, const int* in_sizes, int n_in,
                              void* d_out, int out_size, void* d_ws, size_t ws_size,
                              hipStream_t stream){
  (void)n_in; (void)out_size; (void)ws_size;
  const int*   ids   = (const int*)d_in[0];
  const float* emb_c = (const float*)d_in[1];
  const float* emb_a = (const float*)d_in[2];
  const float* W_ih  = (const float*)d_in[3];
  const float* W_hh  = (const float*)d_in[4];
  const float* b_ih  = (const float*)d_in[5];
  const float* b_hh  = (const float*)d_in[6];
  const float* W_cls = (const float*)d_in[7];
  const float* b_cls = (const float*)d_in[8];
  float* out = (float*)d_out;

  const int B = in_sizes[0] / T_SEQ;   // 8192

  // workspace layout (bf16): Wih | Whh | Wcls(padded) | hs_a | hs_c
  char* ws = (char*)d_ws;
  __bf16* Wih_bf  = (__bf16*)ws;                        // 384*128*2 = 98304 B
  __bf16* Whh_bf  = (__bf16*)(ws + 98304);              // 98304 B
  __bf16* Wcls_bf = (__bf16*)(ws + 196608);             // 112*128*2 = 28672 B
  __bf16* hs_a    = (__bf16*)(ws + 225280);             // B*T*H*2 B
  __bf16* hs_c    = hs_a + (size_t)B * T_SEQ * H_DIM;

  prep_kernel<<<(3 * H_DIM * H_DIM + 255) / 256, 256, 0, stream>>>(
      W_ih, W_hh, W_cls, Wih_bf, Whh_bf, Wcls_bf);

  gru_kernel<<<dim3(B / 16, 2), 256, 0, stream>>>(
      ids, emb_a, emb_c, b_ih, b_hh, Wih_bf, Whh_bf, hs_a, hs_c);

  attn_cls_kernel<<<B, 256, 0, stream>>>(hs_a, hs_c, Wcls_bf, b_cls, out);
}
